// CascadedGroupAttention_49108656063024
// MI455X (gfx1250) — matrix operand order repaired and checked by reading, not compiled, not run on hardware
//
#include <hip/hip_runtime.h>
#include <math.h>

// ---------------------------------------------------------------------------
// CascadedGroupAttention fused kernel for MI455X (gfx1250, wave32, WMMA+TDM).
// One workgroup (256 threads = 8 wave32) per batch element; the entire
// per-batch working set uses the full 320KB WGP LDS.  All WMMA operands but proj_w (read from global) are
// staged in LDS as bf16, K-contiguous (fragment loads are ds_load_b128).
// Bulk input/output transfers go through the Tensor Data Mover
// (TENSOR_LOAD_TO_LDS / TENSOR_STORE_FROM_LDS, TENSORcnt-tracked).
// ---------------------------------------------------------------------------

typedef __attribute__((ext_vector_type(16))) __bf16 v16bf;
typedef __attribute__((ext_vector_type(8)))  float  v8f;
typedef __attribute__((ext_vector_type(4)))  unsigned int v4u;
typedef __attribute__((ext_vector_type(8)))  int v8i;
typedef __attribute__((ext_vector_type(4)))  int v4i;

#define DEVINL __device__ __forceinline__

#if defined(__has_builtin)
#if __has_builtin(__builtin_amdgcn_tensor_load_to_lds) && \
    __has_builtin(__builtin_amdgcn_tensor_store_from_lds) && \
    __has_builtin(__builtin_amdgcn_s_wait_tensorcnt)
#define CGA_HAS_TDM 1
#endif
#endif
#ifndef CGA_HAS_TDM
#define CGA_HAS_TDM 0
#endif

namespace cga {
constexpr int BATCH = 512;
constexpr int HEADS = 4;
constexpr int KD    = 16;     // key_dim
constexpr int DH    = 64;     // per-head value dim d
constexpr int CDIM  = 256;    // DIM
constexpr int QKV   = 96;     // 2*KD + DH
constexpr int RES   = 14;
constexpr int N     = 196;    // tokens
constexpr int NP    = 224;    // m/K-dim padding (7*32) for v and p
constexpr int NROWS = 208;    // n-dim padding (13*16)
constexpr int NT    = 13;     // 13 n-tiles of 16 cover 208 >= 196
constexpr int KS    = 5;
constexpr int NWAVE = 8;

// LDS layout (bytes).  catT lives at the top; the f32 out-stage (200704 B)
// aliases regions [0 .. 188416) which are all dead by projection time.
constexpr int OFF_FEATT = 0;                               // bf16[208][64]   26624
constexpr int OFF_W     = OFF_FEATT + NROWS * DH * 2;      // bf16[96][64]    12288
constexpr int OFF_Q     = OFF_W     + QKV * DH * 2;        // f32 [16][224]   14336
constexpr int OFF_QT    = OFF_Q     + KD * NP * 4;         // bf16[208][16]    6656
constexpr int OFF_KT    = OFF_QT    + NROWS * KD * 2;      // bf16[208][16]    6656
constexpr int OFF_V     = OFF_KT    + NROWS * KD * 2;      // bf16[64][224]   28672
constexpr int OFF_P     = OFF_V     + DH * NP * 2;         // bf16[208][224]  93184  (ends 188416)
constexpr int OFF_OUT   = 0;                               // f32 [256][196] 200704 (alias)
constexpr int OFF_CATT  = 221184;                          // bf16[208][256] 106496
constexpr int SMEM      = 327680;                          // = full 320KB WGP LDS
static_assert(OFF_P + NROWS * NP * 2 <= OFF_CATT, "overlap");
static_assert(OFF_OUT + CDIM * N * 4 <= OFF_CATT, "out-stage overlap");
static_assert(OFF_CATT + NROWS * CDIM * 2 <= SMEM, "smem");
} // namespace cga

// A-fragment (16x32 bf16): lane 0-15 -> M=lane,    elems 0..15 = K0..15
//                          lane 16-31 -> M=lane-16, elems 0..15 = K16..31
template <typename F>
DEVINL v16bf load_a_frag(int lane, F f) {
  const int r  = lane & 15;
  const int kb = (lane >> 4) << 4;
  v16bf t;
#pragma unroll
  for (int j = 0; j < 16; ++j) t[j] = f(r, kb + j);
  return t;
}

// B-fragment (32x16 bf16): lane 0-15 -> N=lane, elems 0..15 = K0..15
//                          lane 16-31 -> N=lane-16, elems 0..15 = K16..31
template <typename F>
DEVINL v16bf load_b_frag(int lane, F f) {
  const int r  = lane & 15;
  const int kb = (lane >> 4) << 4;
  v16bf t;
#pragma unroll
  for (int j = 0; j < 16; ++j) t[j] = f(r, kb + j);
  return t;
}

DEVINL v8f wmma_bf16(v16bf a, v16bf b, v8f c) {
  return __builtin_amdgcn_wmma_f32_16x16x32_bf16(false, a, false, b, (short)0, c,
                                                 false, false);
}

#if CGA_HAS_TDM
// Build a 1-D TDM descriptor (D# groups 0/1) for nelem contiguous f32.
// Group0: count=1 | lds_addr | global_addr[56:0] | type=2 ("image").
// Group1: data_size=4B (code 2); tensor_dim0 = tile_dim0 = dim0_stride = nelem.
DEVINL void tdm_desc_1d(v4u& g0, v8i& g1, unsigned lds_off, const void* gptr,
                        unsigned nelem) {
  const unsigned long long ga = (unsigned long long)gptr;
  g0 = v4u{1u, lds_off, (unsigned)(ga & 0xffffffffull),
           (unsigned)((ga >> 32) & 0x01ffffffull) | 0x80000000u};
  g1 = v8i{(int)(2u << 16),                       // data_size = 4 bytes
           (int)((nelem & 0xffffu) << 16),        // tensor_dim0[15:0]  @ bits 63:48
           (int)((nelem >> 16) & 0xffffu),        // tensor_dim0[31:16]; dim1=0
           (int)((nelem & 0xffffu) << 16),        // tile_dim0 @ bits 127:112
           0,                                     // tile_dim1 = tile_dim2 = 0
           (int)nelem,                            // tensor_dim0_stride[31:0]
           0, 0};
}

DEVINL void tdm_load_1d(unsigned lds_off, const void* gptr, unsigned nelem) {
  v4u g0; v8i g1; tdm_desc_1d(g0, g1, lds_off, gptr, nelem);
  const v4i z4 = {0, 0, 0, 0};
#if defined(__clang_major__) && __clang_major__ >= 23
  const v8i z8 = {0, 0, 0, 0, 0, 0, 0, 0};
  __builtin_amdgcn_tensor_load_to_lds(g0, g1, z4, z4, z8, 0);
#else
  __builtin_amdgcn_tensor_load_to_lds(g0, g1, z4, z4, 0);
#endif
}

DEVINL void tdm_store_1d(unsigned lds_off, void* gptr, unsigned nelem) {
  v4u g0; v8i g1; tdm_desc_1d(g0, g1, lds_off, gptr, nelem);
  const v4i z4 = {0, 0, 0, 0};
#if defined(__clang_major__) && __clang_major__ >= 23
  const v8i z8 = {0, 0, 0, 0, 0, 0, 0, 0};
  __builtin_amdgcn_tensor_store_from_lds(g0, g1, z4, z4, z8, 0);
#else
  __builtin_amdgcn_tensor_store_from_lds(g0, g1, z4, z4, 0);
#endif
}
#endif // CGA_HAS_TDM

extern "C" __global__ __launch_bounds__(256)
void cga_kernel(const float* __restrict__ x,
                const float* __restrict__ qkv_w,
                const float* __restrict__ qkv_scale,
                const float* __restrict__ qkv_bias,
                const float* __restrict__ dw_w,
                const float* __restrict__ dw_scale,
                const float* __restrict__ dw_bias,
                const float* __restrict__ proj_w,
                const float* __restrict__ proj_scale,
                const float* __restrict__ proj_bias,
                const float* __restrict__ attention_biases,
                const int*   __restrict__ bias_idxs,
                float* __restrict__ out) {
  using namespace cga;
  extern __shared__ char smem[];
  __bf16* s_featT = reinterpret_cast<__bf16*>(smem + OFF_FEATT); // [n][c] c-contig
  __bf16* s_w     = reinterpret_cast<__bf16*>(smem + OFF_W);     // [o][c] c-contig
  float*  s_q     = reinterpret_cast<float*>(smem + OFF_Q);      // [c][n] (dwconv)
  __bf16* s_qT    = reinterpret_cast<__bf16*>(smem + OFF_QT);    // [n][c] c-contig
  __bf16* s_kT    = reinterpret_cast<__bf16*>(smem + OFF_KT);    // [m][c] c-contig
  __bf16* s_v     = reinterpret_cast<__bf16*>(smem + OFF_V);     // [d][m] m-contig
  __bf16* s_p     = reinterpret_cast<__bf16*>(smem + OFF_P);     // [n][m] m-contig
  __bf16* s_catT  = reinterpret_cast<__bf16*>(smem + OFF_CATT);  // [n][c] c-contig
  float*  s_dq    = reinterpret_cast<float*>(smem + OFF_P);      // alias: p not live
  float*  s_xs    = reinterpret_cast<float*>(smem + OFF_P);      // alias: x stage
  float*  s_out   = reinterpret_cast<float*>(smem + OFF_OUT);    // alias: out stage

  const int tid  = threadIdx.x;
  const int lane = tid & 31;
  const int wv   = tid >> 5;
  const int b    = blockIdx.x;
  const float* xb = x + (size_t)b * CDIM * N;

  // Pull projection weights (reused by every workgroup) toward cache early.
  for (int off = tid * 256; off < CDIM * CDIM * 4; off += 256 * 256 * 4)
    __builtin_prefetch(reinterpret_cast<const char*>(proj_w) + off, 0, 3);

#if CGA_HAS_TDM
  // DMA head-0's x slice (64x196 f32, contiguous) into the p-region stage.
  if (wv == 0) {
    tdm_load_1d(OFF_P, xb, DH * N);
    __builtin_amdgcn_s_wait_tensorcnt(0);
  }
#endif

  for (int i = 0; i < HEADS; ++i) {
    // ---- stage per-head qkv weights as bf16 (K-contiguous for A-frags) ----
    for (int idx = tid; idx < QKV * DH; idx += 256)
      s_w[idx] = (__bf16)qkv_w[i * QKV * DH + idx];
    // ---- head 0: build featT; later heads get it from the AV scatter ----
    if (i == 0) {
      __syncthreads();  // TDM stage visible (wave0 waited before the barrier)
      for (int idx = tid; idx < NROWS * DH; idx += 256) {
        const int n = idx / DH, c = idx % DH;
#if CGA_HAS_TDM
        s_featT[idx] = (n < N) ? (__bf16)s_xs[c * N + n] : (__bf16)0.f;
#else
        s_featT[idx] = (n < N) ? (__bf16)xb[c * N + n] : (__bf16)0.f;
#endif
      }
    }
    __syncthreads();

    // ---- QKV: y[96xN] = W(96x64) @ feat(64xN); affine; split q/kT/v ----
    {
      const float* sc = qkv_scale + i * QKV;
      const float* bs = qkv_bias  + i * QKV;
      for (int t = wv; t < 6 * NT; t += NWAVE) {
        const int mt = t / NT, nt = t % NT;
        v8f acc = {};
#pragma unroll
        for (int kk = 0; kk < 2; ++kk) {
          v16bf af = load_a_frag(lane, [&](int r, int k) -> __bf16 {
            return s_w[(mt * 16 + r) * DH + kk * 32 + k]; });
          v16bf bf = load_b_frag(lane, [&](int r, int k) -> __bf16 {
            return s_featT[(nt * 16 + r) * DH + kk * 32 + k]; });
          acc = wmma_bf16(af, bf, acc);
        }
        const int nn = nt * 16 + (lane & 15);
        const int mh = (lane >> 4) << 3;
        if (mt == 0) {           // rows 0..15 -> q (f32, spatial layout)
#pragma unroll
          for (int r = 0; r < 8; ++r) {
            const int o = mh + r;
            s_q[o * NP + nn] = acc[r] * sc[o] + bs[o];
          }
        } else if (mt == 1) {    // rows 16..31 -> k^T (contiguous b16 x8)
#pragma unroll
          for (int r = 0; r < 8; ++r) {
            const int o = 16 + mh + r;
            s_kT[nn * KD + mh + r] = (__bf16)(acc[r] * sc[o] + bs[o]);
          }
        } else {                 // rows 32..95 -> v [d][m]
#pragma unroll
          for (int r = 0; r < 8; ++r) {
            const int o = mt * 16 + mh + r;
            s_v[(o - 2 * KD) * NP + nn] = (__bf16)(acc[r] * sc[o] + bs[o]);
          }
        }
      }
    }
    __syncthreads();

    // ---- depthwise 5x5 on q, affine, exact GELU, residual -> qT (bf16) ----
    {
      const float* dwi = dw_w    + i * KD * KS * KS;
      const float* dsc = dw_scale + i * KD;
      const float* dbs = dw_bias  + i * KD;
      for (int idx = tid; idx < KD * N; idx += 256) {
        const int c = idx / N, n = idx % N, yy = n / RES, xx = n % RES;
        float sum = 0.f;
#pragma unroll
        for (int ky = 0; ky < KS; ++ky) {
          const int iy = yy + ky - 2;
          if (iy < 0 || iy >= RES) continue;
#pragma unroll
          for (int kx = 0; kx < KS; ++kx) {
            const int ix = xx + kx - 2;
            if (ix < 0 || ix >= RES) continue;
            sum += s_q[c * NP + iy * RES + ix] * dwi[c * KS * KS + ky * KS + kx];
          }
        }
        s_dq[idx] = sum * dsc[c] + dbs[c];
      }
      __syncthreads();
      for (int idx = tid; idx < KD * N; idx += 256) {
        const int c = idx / N, n = idx % N;
        float g = s_dq[idx];
        g = 0.5f * g * (1.f + erff(g * 0.70710678118f));
        s_qT[n * KD + c] = (__bf16)(g + s_q[c * NP + n]);
      }
      __syncthreads();
    }

    // ---- logits: attn[n][m] = (q^T k)*0.25 + bias_gather; K=16 pad to 32 ----
    {
      const float* ab = attention_biases + i * N;
      for (int t = wv; t < NT * NT; t += NWAVE) {
        const int qt = t / NT, kt = t % NT;
        v8f acc = {};
        v16bf af = load_a_frag(lane, [&](int r, int k) -> __bf16 {
          return (k < KD) ? s_qT[(qt * 16 + r) * KD + k] : (__bf16)0.f; });
        v16bf bf = load_b_frag(lane, [&](int r, int k) -> __bf16 {
          return (k < KD) ? s_kT[(kt * 16 + r) * KD + k] : (__bf16)0.f; });
        acc = wmma_bf16(af, bf, acc);
        const int m  = kt * 16 + (lane & 15);
        const int qh = (lane >> 4) << 3;
#pragma unroll
        for (int r = 0; r < 8; ++r) {
          const int n = qt * 16 + qh + r;
          float lg = acc[r] * 0.25f;
          if (n < N && m < N) lg += ab[bias_idxs[n * N + m]];
          s_p[n * NP + m] = (__bf16)lg;
        }
      }
    }
    __syncthreads();

    // ---- row softmax; zero all n/m padding so padded WMMA K-slices vanish ----
    for (int n = tid; n < NROWS; n += 256) {
      __bf16* row = s_p + n * NP;
      if (n < N) {
        float mx = -1e30f;
        for (int m = 0; m < N; ++m) mx = fmaxf(mx, (float)row[m]);
        float sum = 0.f;
        for (int m = 0; m < N; ++m) {
          const float e = __expf((float)row[m] - mx);
          sum += e;
          row[m] = (__bf16)e;
        }
        const float inv = 1.f / sum;
        for (int m = 0; m < N; ++m) row[m] = (__bf16)((float)row[m] * inv);
        for (int m = N; m < NP; ++m) row[m] = (__bf16)0.f;
      } else {
        for (int m = 0; m < NP; ++m) row[m] = (__bf16)0.f;
      }
    }
    __syncthreads();

    // ---- AV: out[d][n] = sum_m v[d][m] p[n][m]; scatter fuses relu->catT
    //      and the cascade (out + next x slice) -> featT, both b128 stores ----
    for (int t = wv; t < 4 * NT; t += NWAVE) {
      const int mt = t / NT, nt = t % NT;
      v8f acc = {};
#pragma unroll
      for (int kk = 0; kk < 7; ++kk) {
        v16bf af = load_a_frag(lane, [&](int r, int k) -> __bf16 {
          return s_v[(mt * 16 + r) * NP + kk * 32 + k]; });
        v16bf bf = load_b_frag(lane, [&](int r, int k) -> __bf16 {
          return s_p[(nt * 16 + r) * NP + kk * 32 + k]; });
        acc = wmma_bf16(af, bf, acc);
      }
      const int nn = nt * 16 + (lane & 15);
      const int mh = (lane >> 4) << 3;
#pragma unroll
      for (int r = 0; r < 8; ++r) {
        const int dd = mt * 16 + mh + r;
        const float val = acc[r];                      // == 0 for nn >= N
        s_catT[nn * CDIM + i * DH + dd] = (__bf16)fmaxf(val, 0.f);
        if (i < HEADS - 1) {
          const float xn = (nn < N) ? xb[((i + 1) * DH + dd) * N + nn] : 0.f;
          s_featT[nn * DH + dd] = (__bf16)(val + xn);
        }
      }
    }
    __syncthreads();
  }

  // ---- projection: out[o][n] = affine( proj_w(256x256) @ cat(256xN) );
  //      staged in LDS, then written back in bulk ----
  for (int t = wv; t < 16 * NT; t += NWAVE) {
    const int mt = t / NT, nt = t % NT;
    v8f acc = {};
#pragma unroll
    for (int kk = 0; kk < 8; ++kk) {
      v16bf af = load_a_frag(lane, [&](int r, int k) -> __bf16 {
        return (__bf16)proj_w[(mt * 16 + r) * CDIM + kk * 32 + k]; });
      v16bf bf = load_b_frag(lane, [&](int r, int k) -> __bf16 {
        return s_catT[(nt * 16 + r) * CDIM + kk * 32 + k]; });
      acc = wmma_bf16(af, bf, acc);
    }
    const int nn = nt * 16 + (lane & 15);
    const int mh = (lane >> 4) << 3;
    if (nn < N) {
#pragma unroll
      for (int r = 0; r < 8; ++r) {
        const int o = mt * 16 + mh + r;
        s_out[o * N + nn] = acc[r] * proj_scale[o] + proj_bias[o];
      }
    }
  }
  __syncthreads();

#if CGA_HAS_TDM
  // Bulk writeback of the contiguous 256x196 f32 result via the TDM.
  if (wv == 0) {
    tdm_store_1d(OFF_OUT, out + (size_t)b * CDIM * N, CDIM * N);
    __builtin_amdgcn_s_wait_tensorcnt(0);
  }
#else
  {
    float4* dst = reinterpret_cast<float4*>(out + (size_t)b * CDIM * N);
    const float4* src = reinterpret_cast<const float4*>(s_out);
    for (int idx = tid; idx < CDIM * N / 4; idx += 256) dst[idx] = src[idx];
  }
#endif
}

extern "C" void kernel_launch(void* const* d_in, const int* in_sizes, int n_in,
                              void* d_out, int out_size, void* d_ws, size_t ws_size,
                              hipStream_t stream) {
  using namespace cga;
  const float* x    = (const float*)d_in[0];
  const float* qw   = (const float*)d_in[1];
  const float* qs   = (const float*)d_in[2];
  const float* qb   = (const float*)d_in[3];
  const float* dww  = (const float*)d_in[4];
  const float* dws  = (const float*)d_in[5];
  const float* dwb  = (const float*)d_in[6];
  const float* pw   = (const float*)d_in[7];
  const float* ps   = (const float*)d_in[8];
  const float* pb   = (const float*)d_in[9];
  const float* ab   = (const float*)d_in[10];
  const int*   bidx = (const int*)d_in[11];
  float* outp = (float*)d_out;

  // Full 320KB WGP LDS per workgroup; opt in explicitly.
  (void)hipFuncSetAttribute(reinterpret_cast<const void*>(cga_kernel),
                            hipFuncAttributeMaxDynamicSharedMemorySize, SMEM);
  cga_kernel<<<BATCH, 256, SMEM, stream>>>(x, qw, qs, qb, dww, dws, dwb,
                                           pw, ps, pb, ab, bidx, outp);
}
